// FlashHunyuanDecoderLayer_47356309405792
// MI455X (gfx1250) — compile-verified, matrix operand order checked by reading, not run on hardware
//
#include <hip/hip_runtime.h>
#include <cstdint>
#include <cstddef>

typedef unsigned short bh16;
typedef __attribute__((ext_vector_type(16))) __bf16 v16bf;
typedef __attribute__((ext_vector_type(8)))  float  v8f;

#define T_  2048
#define H_  2048
#define NH_ 16
#define HD_ 128
#define E_  16
#define I_  2048
#define IS_ 2048
#define EPSI 1e-6f

__device__ __forceinline__ bh16 f2bf(float f) {
  union { float f; unsigned u; } a; a.f = f;
  unsigned r = a.u + 0x7FFFu + ((a.u >> 16) & 1u);   // round-to-nearest-even
  return (bh16)(r >> 16);
}
__device__ __forceinline__ float bf2f(bh16 s) {
  union { unsigned u; float f; } a; a.u = ((unsigned)s) << 16; return a.f;
}
__device__ __forceinline__ __bf16 bfc(bh16 s) {
  union { bh16 u; __bf16 b; } a; a.u = s; return a.b;
}
__device__ __forceinline__ v8f vzero8() {
  v8f z = {0.f, 0.f, 0.f, 0.f, 0.f, 0.f, 0.f, 0.f}; return z;
}
// Low 32 bits of a generic pointer to LDS = LDS byte offset (ISA 10.2).
__device__ __forceinline__ unsigned lds_off(const void* p) {
  return (unsigned)(unsigned long long)p;
}
// CDNA5 async copy global -> LDS (16B per lane), tracked on ASYNCcnt.
__device__ __forceinline__ void async_b128(unsigned lds, const bh16* g) {
  asm volatile("global_load_async_to_lds_b128 %0, %1, off"
               :: "v"(lds), "v"(g) : "memory");
}
__device__ __forceinline__ void wait_async0() {
  asm volatile("s_wait_asynccnt 0x0" ::: "memory");
}

// A-fragment (16-bit, 16x32): lanes 0-15 hold K {0..7,16..23}, lanes 16-31 hold K {8..15,24..31}.
// B-fragment mirrors A with the 16 output columns in lane%16.
__device__ __forceinline__ v16bf frag_row(const bh16* p, int hs) {
  v16bf f;
#pragma unroll
  for (int j = 0; j < 8; j++) { f[j] = bfc(p[hs + j]); f[j + 8] = bfc(p[16 + hs + j]); }
  return f;
}
__device__ __forceinline__ v16bf frag_col(const bh16* p, int stride, int hs) {
  v16bf f;
#pragma unroll
  for (int j = 0; j < 8; j++) { f[j] = bfc(p[(hs + j) * stride]); f[j + 8] = bfc(p[(16 + hs + j) * stride]); }
  return f;
}

#define WMMA_BF16(a, b, c) \
  __builtin_amdgcn_wmma_f32_16x16x32_bf16(false, (a), false, (b), (short)0, (c), false, false)

// ---------------------------------------------------------------------------
// fp32 -> bf16 bulk weight conversion (one pass at layer start)
// ---------------------------------------------------------------------------
__global__ __launch_bounds__(256) void k_cvt(const float* __restrict__ s,
                                             bh16* __restrict__ d, long n) {
  long i = ((long)blockIdx.x * 256 + threadIdx.x) * 4;
  if (i + 3 < n) {
    float4 f = *(const float4*)(s + i);
    d[i]     = f2bf(f.x);
    d[i + 1] = f2bf(f.y);
    d[i + 2] = f2bf(f.z);
    d[i + 3] = f2bf(f.w);
  }
}

// ---------------------------------------------------------------------------
// RMSNorm: fp32 row -> bf16 row
// ---------------------------------------------------------------------------
__global__ __launch_bounds__(256) void k_rms(const float* __restrict__ x,
                                             const float* __restrict__ w,
                                             bh16* __restrict__ o) {
  int t = blockIdx.x, tid = threadIdx.x;
  __shared__ float red[256];
  __shared__ float sc;
  const float* xr = x + (size_t)t * H_;
  float ss = 0.f;
  for (int h = tid; h < H_; h += 256) { float v = xr[h]; ss += v * v; }
  red[tid] = ss; __syncthreads();
  for (int s = 128; s > 0; s >>= 1) { if (tid < s) red[tid] += red[tid + s]; __syncthreads(); }
  if (tid == 0) sc = rsqrtf(red[0] / (float)H_ + EPSI);
  __syncthreads();
  bh16* orow = o + (size_t)t * H_;
  for (int h = tid; h < H_; h += 256) orow[h] = f2bf(xr[h] * sc * w[h]);
}

// ---------------------------------------------------------------------------
// WMMA GEMM: C[M,N] = A_bf16[M,K] @ W[N,K]^T  (+ optional residual)
// W is bf16 (async-copied straight to LDS) or fp32 (converted in-flight).
// 64x64 block tile, 4 waves, each wave 2x2 of 16x16 WMMA tiles, K step 32.
// ---------------------------------------------------------------------------
template <bool WBF16>
__global__ __launch_bounds__(128) void k_gemm(const bh16* __restrict__ A,
                                              const void* __restrict__ Wv,
                                              float* __restrict__ C,
                                              const float* __restrict__ R,
                                              int M, int N, int K) {
  __shared__ __align__(16) bh16 As[64][40];
  __shared__ __align__(16) bh16 Bs[64][40];
  int tid = threadIdx.x, lane = tid & 31, wv = tid >> 5;
  int bm = blockIdx.y * 64, bn = blockIdx.x * 64;
  int wm = (wv >> 1) * 32, wn = (wv & 1) * 32;
  int r = lane & 15, hs = (lane < 16) ? 0 : 8;
  v8f acc[2][2] = {{vzero8(), vzero8()}, {vzero8(), vzero8()}};

  int lrow = tid >> 1, lhk = (tid & 1) * 16;
  const bh16* Arow = A + (size_t)(bm + lrow) * K + lhk;
  unsigned ldsA = lds_off(&As[lrow][lhk]);
  unsigned ldsB = lds_off(&Bs[lrow][lhk]);
  const bh16*  Wbh = nullptr;
  const float* Wfl = nullptr;
  if constexpr (WBF16) Wbh = (const bh16*)Wv + (size_t)(bn + lrow) * K + lhk;
  else                 Wfl = (const float*)Wv + (size_t)(bn + lrow) * K + lhk;

  for (int k0 = 0; k0 < K; k0 += 32) {
    async_b128(ldsA,      Arow + k0);
    async_b128(ldsA + 16, Arow + k0 + 8);
    if constexpr (WBF16) {
      async_b128(ldsB,      Wbh + k0);
      async_b128(ldsB + 16, Wbh + k0 + 8);
      if (k0 + 32 < K) {
        __builtin_prefetch(Arow + k0 + 32, 0, 0);
        __builtin_prefetch(Wbh + k0 + 32, 0, 0);
      }
    } else {
      const float4* s4 = (const float4*)(Wfl + k0);
      float4 f0 = s4[0], f1 = s4[1], f2 = s4[2], f3 = s4[3];
      bh16* dst = &Bs[lrow][lhk];
      dst[0]  = f2bf(f0.x); dst[1]  = f2bf(f0.y); dst[2]  = f2bf(f0.z); dst[3]  = f2bf(f0.w);
      dst[4]  = f2bf(f1.x); dst[5]  = f2bf(f1.y); dst[6]  = f2bf(f1.z); dst[7]  = f2bf(f1.w);
      dst[8]  = f2bf(f2.x); dst[9]  = f2bf(f2.y); dst[10] = f2bf(f2.z); dst[11] = f2bf(f2.w);
      dst[12] = f2bf(f3.x); dst[13] = f2bf(f3.y); dst[14] = f2bf(f3.z); dst[15] = f2bf(f3.w);
    }
    wait_async0();
    __syncthreads();
    v16bf af0 = frag_row(&As[wm + r][0], hs);
    v16bf af1 = frag_row(&As[wm + 16 + r][0], hs);
    v16bf bf0 = frag_row(&Bs[wn + r][0], hs);
    v16bf bf1 = frag_row(&Bs[wn + 16 + r][0], hs);
    acc[0][0] = WMMA_BF16(af0, bf0, acc[0][0]);
    acc[0][1] = WMMA_BF16(af0, bf1, acc[0][1]);
    acc[1][0] = WMMA_BF16(af1, bf0, acc[1][0]);
    acc[1][1] = WMMA_BF16(af1, bf1, acc[1][1]);
    __syncthreads();
  }
#pragma unroll
  for (int mi = 0; mi < 2; mi++)
#pragma unroll
    for (int ni = 0; ni < 2; ni++)
#pragma unroll
      for (int e = 0; e < 8; e++) {
        size_t row = (size_t)(bm + wm + mi * 16 + hs + e);
        size_t col = (size_t)(bn + wn + ni * 16 + r);
        float v = acc[mi][ni][e];
        if (R) v += R[row * (size_t)N + col];
        C[row * (size_t)N + col] = v;
      }
}

// ---------------------------------------------------------------------------
// QKV post-process: per (token, head) QK-RMSNorm + RoPE, emit bf16 q/k/v.
// ---------------------------------------------------------------------------
__global__ __launch_bounds__(128) void k_qkvpost(const float* __restrict__ qkv,
                                                 const float* __restrict__ qw,
                                                 const float* __restrict__ kw,
                                                 bh16* __restrict__ qb,
                                                 bh16* __restrict__ kb,
                                                 bh16* __restrict__ vb) {
  int t = blockIdx.x, head = blockIdx.y, tid = threadIdx.x;
  __shared__ float red[128];
  __shared__ float qn[128], kn[128];
  __shared__ float scq, sck;
  size_t base = (size_t)t * (3 * H_) + (size_t)head * HD_;
  float qv = qkv[base + tid];
  float kv = qkv[base + H_ + tid];
  float vv = qkv[base + 2 * H_ + tid];

  red[tid] = qv * qv; __syncthreads();
  for (int s = 64; s > 0; s >>= 1) { if (tid < s) red[tid] += red[tid + s]; __syncthreads(); }
  if (tid == 0) scq = rsqrtf(red[0] / (float)HD_ + EPSI);
  __syncthreads();
  red[tid] = kv * kv; __syncthreads();
  for (int s = 64; s > 0; s >>= 1) { if (tid < s) red[tid] += red[tid + s]; __syncthreads(); }
  if (tid == 0) sck = rsqrtf(red[0] / (float)HD_ + EPSI);
  __syncthreads();

  qn[tid] = qv * scq * qw[tid];
  kn[tid] = kv * sck * kw[tid];
  __syncthreads();

  size_t ob = ((size_t)t * NH_ + head) * HD_;
  vb[ob + tid] = f2bf(vv);

  int d = tid & 63;
  float inv = __expf(-((float)(2 * d) / (float)HD_) * __logf(10000.0f));
  float ang = (float)t * inv;
  float cs = __cosf(ang), sn = __sinf(ang);
  if (tid < 64) {
    float t1 = qn[2 * d], t2 = qn[2 * d + 1];
    qb[ob + 2 * d]     = f2bf(t1 * cs - t2 * sn);
    qb[ob + 2 * d + 1] = f2bf(t1 * sn + t2 * cs);
  } else {
    float t1 = kn[2 * d], t2 = kn[2 * d + 1];
    kb[ob + 2 * d]     = f2bf(t1 * cs - t2 * sn);
    kb[ob + 2 * d + 1] = f2bf(t1 * sn + t2 * cs);
  }
}

// ---------------------------------------------------------------------------
// Flash attention: one wave per (head, 16-query tile); causal, online softmax.
// K/V chunks staged via async global->LDS copies.
// ---------------------------------------------------------------------------
__global__ __launch_bounds__(32) void k_attn(const bh16* __restrict__ qb,
                                             const bh16* __restrict__ kb,
                                             const bh16* __restrict__ vb,
                                             bh16* __restrict__ ob) {
  int head = blockIdx.x;
  int q0 = blockIdx.y * 16;
  int lane = threadIdx.x;
  int r = lane & 15, hs = (lane < 16) ? 0 : 8;

  __shared__ __align__(16) bh16 Kt[32 * HD_];
  __shared__ __align__(16) bh16 Vt[32 * HD_];
  __shared__ __align__(16) bh16 Pt[16 * 32];

  const bh16* qrow = qb + ((size_t)(q0 + r) * NH_ + head) * HD_;
  v16bf qf[4];
#pragma unroll
  for (int kc = 0; kc < 4; kc++) qf[kc] = frag_row(qrow + kc * 32, hs);

  v8f of[8];
#pragma unroll
  for (int hi = 0; hi < 8; hi++) of[hi] = vzero8();
  float mrow[8], lrow[8];
#pragma unroll
  for (int e = 0; e < 8; e++) { mrow[e] = -3.0e38f; lrow[e] = 0.f; }

  unsigned ldsK = lds_off(Kt + lane * HD_);
  unsigned ldsV = lds_off(Vt + lane * HD_);
  const float scale = 0.08838834764831845f;   // 1/sqrt(128)
  int kmax = (q0 + 15) >> 5;
  for (int kbi = 0; kbi <= kmax; kbi++) {
    int kbase = kbi * 32;
    {   // async-stage K,V chunk (each lane copies its own 256B row)
      const bh16* krow = kb + ((size_t)(kbase + lane) * NH_ + head) * HD_;
      const bh16* vrow = vb + ((size_t)(kbase + lane) * NH_ + head) * HD_;
#pragma unroll
      for (int j = 0; j < 16; j++) {
        async_b128(ldsK + j * 16, krow + j * 8);
        async_b128(ldsV + j * 16, vrow + j * 8);
      }
    }
    wait_async0();
    __syncthreads();

    v8f sf[2];
#pragma unroll
    for (int ni = 0; ni < 2; ni++) {
      v8f s = vzero8();
#pragma unroll
      for (int kc = 0; kc < 4; kc++) {
        v16bf bfK = frag_row(Kt + (ni * 16 + r) * HD_ + kc * 32, hs);
        s = WMMA_BF16(qf[kc], bfK, s);
      }
      sf[ni] = s;
    }
    // scale + causal mask
#pragma unroll
    for (int ni = 0; ni < 2; ni++) {
      int colg = kbase + ni * 16 + r;
#pragma unroll
      for (int e = 0; e < 8; e++) {
        int rowg = q0 + hs + e;
        float v = sf[ni][e] * scale;
        if (colg > rowg) v = -3.0e38f;
        sf[ni][e] = v;
      }
    }
    // online softmax stats (rows live across the 16-lane group)
    float mnew[8], alpha[8];
#pragma unroll
    for (int e = 0; e < 8; e++) {
      float rm = fmaxf(sf[0][e], sf[1][e]);
      rm = fmaxf(rm, __shfl_xor(rm, 1, 16));
      rm = fmaxf(rm, __shfl_xor(rm, 2, 16));
      rm = fmaxf(rm, __shfl_xor(rm, 4, 16));
      rm = fmaxf(rm, __shfl_xor(rm, 8, 16));
      mnew[e] = fmaxf(mrow[e], rm);
      alpha[e] = __expf(mrow[e] - mnew[e]);
      mrow[e] = mnew[e];
    }
#pragma unroll
    for (int ni = 0; ni < 2; ni++)
#pragma unroll
      for (int e = 0; e < 8; e++) {
        float p = __expf(sf[ni][e] - mnew[e]);
        sf[ni][e] = p;
        Pt[(hs + e) * 32 + ni * 16 + r] = f2bf(p);
      }
#pragma unroll
    for (int e = 0; e < 8; e++) {
      float rs = sf[0][e] + sf[1][e];
      rs += __shfl_xor(rs, 1, 16);
      rs += __shfl_xor(rs, 2, 16);
      rs += __shfl_xor(rs, 4, 16);
      rs += __shfl_xor(rs, 8, 16);
      lrow[e] = lrow[e] * alpha[e] + rs;
    }
#pragma unroll
    for (int hi = 0; hi < 8; hi++)
#pragma unroll
      for (int e = 0; e < 8; e++) of[hi][e] *= alpha[e];
    __syncthreads();

    v16bf pf = frag_row(Pt + r * 32, hs);
#pragma unroll
    for (int hi = 0; hi < 8; hi++) {
      v16bf vf = frag_col(Vt + hi * 16 + r, HD_, hs);
      of[hi] = WMMA_BF16(pf, vf, of[hi]);
    }
    __syncthreads();
  }
#pragma unroll
  for (int hi = 0; hi < 8; hi++)
#pragma unroll
    for (int e = 0; e < 8; e++) {
      size_t row = (size_t)(q0 + hs + e);
      ob[row * H_ + head * HD_ + hi * 16 + r] = f2bf(of[hi][e] / lrow[e]);
    }
}

// ---------------------------------------------------------------------------
// MoE router: top-1 of 16 experts, softmax prob of the winner.
// ---------------------------------------------------------------------------
__global__ __launch_bounds__(128) void k_router(const bh16* __restrict__ h2b,
                                                const float* __restrict__ gw,
                                                int* __restrict__ topi,
                                                float* __restrict__ topw) {
  int t = blockIdx.x, tid = threadIdx.x;
  __shared__ float red[128];
  __shared__ float logits[16];
  const bh16* hrow = h2b + (size_t)t * H_;
  for (int e = 0; e < E_; e++) {
    float acc = 0.f;
    const float* grow = gw + (size_t)e * H_;
    for (int h = tid; h < H_; h += 128) acc += bf2f(hrow[h]) * grow[h];
    red[tid] = acc; __syncthreads();
    for (int s = 64; s > 0; s >>= 1) { if (tid < s) red[tid] += red[tid + s]; __syncthreads(); }
    if (tid == 0) logits[e] = red[0];
    __syncthreads();
  }
  if (tid == 0) {
    float mx = logits[0]; int bi = 0;
    for (int e = 1; e < E_; e++) if (logits[e] > mx) { mx = logits[e]; bi = e; }
    float ssum = 0.f;
    for (int e = 0; e < E_; e++) ssum += __expf(logits[e] - mx);
    topi[t] = bi; topw[t] = 1.0f / ssum;
  }
}

__global__ void k_zero16(int* __restrict__ c) { if (threadIdx.x < 16) c[threadIdx.x] = 0; }
__global__ void k_count(const int* __restrict__ topi, int* __restrict__ counts) {
  int t = blockIdx.x * 256 + threadIdx.x;
  if (t < T_) atomicAdd(&counts[topi[t]], 1);
}
__global__ void k_scan(const int* __restrict__ counts, int* __restrict__ offs, int* __restrict__ cursor) {
  if (threadIdx.x == 0) {
    offs[0] = 0;
    for (int e = 0; e < E_; e++) { offs[e + 1] = offs[e] + counts[e]; cursor[e] = 0; }
  }
}
__global__ void k_scatter(const int* __restrict__ topi, const int* __restrict__ offs,
                          int* __restrict__ cursor, int* __restrict__ sorted) {
  int t = blockIdx.x * 256 + threadIdx.x;
  if (t < T_) { int e = topi[t]; int p = atomicAdd(&cursor[e], 1); sorted[offs[e] + p] = t; }
}

// ---------------------------------------------------------------------------
// MoE gate/up gather-GEMM: gu[pos, 4096] = h2[tok(pos)] @ W_e^T
// Rows are index-clamped (never OOB); results masked at the store.
// ---------------------------------------------------------------------------
template <bool WBF16>
__global__ __launch_bounds__(128) void k_moe_gu(const bh16* __restrict__ h2b,
                                                const void* __restrict__ eguv,
                                                const int* __restrict__ sorted,
                                                const int* __restrict__ offs,
                                                float* __restrict__ gu) {
  int e = blockIdx.z, mt = blockIdx.y, nt = blockIdx.x;
  int seg = offs[e], cnt = offs[e + 1] - seg;
  if (mt * 16 >= cnt) return;
  __shared__ int tok[16];
  __shared__ __align__(16) bh16 As[16][40];
  __shared__ __align__(16) bh16 Bs[64][40];
  int tid = threadIdx.x, lane = tid & 31, wv = tid >> 5;
  int r = lane & 15, hs = (lane < 16) ? 0 : 8;
  if (tid < 16) {
    int idx = mt * 16 + tid; if (idx >= cnt) idx = cnt - 1;
    tok[tid] = sorted[seg + idx];
  }
  __syncthreads();

  const bh16*  Wbh = nullptr;
  const float* Wfl = nullptr;
  int lrow = tid >> 1, lhk = (tid & 1) * 16;
  if constexpr (WBF16) Wbh = (const bh16*)eguv + (size_t)e * (2 * I_) * H_ + (size_t)(nt * 64 + lrow) * H_ + lhk;
  else                 Wfl = (const float*)eguv + (size_t)e * (2 * I_) * H_ + (size_t)(nt * 64 + lrow) * H_ + lhk;

  int arr = tid >> 2, asg = (tid & 3) * 8;           // A-tile async mapping (tid<64)
  const bh16* Agr = h2b + (size_t)tok[arr & 15] * H_ + asg;
  unsigned ldsAg = lds_off(&As[arr & 15][asg]);
  unsigned ldsB = lds_off(&Bs[lrow][lhk]);

  v8f acc = vzero8();
  for (int k0 = 0; k0 < H_; k0 += 32) {
    if (tid < 64) async_b128(ldsAg, Agr + k0);
    if constexpr (WBF16) {
      async_b128(ldsB,      Wbh + k0);
      async_b128(ldsB + 16, Wbh + k0 + 8);
    } else {
      const float4* s4 = (const float4*)(Wfl + k0);
      float4 f0 = s4[0], f1 = s4[1], f2 = s4[2], f3 = s4[3];
      bh16* dst = &Bs[lrow][lhk];
      dst[0]  = f2bf(f0.x); dst[1]  = f2bf(f0.y); dst[2]  = f2bf(f0.z); dst[3]  = f2bf(f0.w);
      dst[4]  = f2bf(f1.x); dst[5]  = f2bf(f1.y); dst[6]  = f2bf(f1.z); dst[7]  = f2bf(f1.w);
      dst[8]  = f2bf(f2.x); dst[9]  = f2bf(f2.y); dst[10] = f2bf(f2.z); dst[11] = f2bf(f2.w);
      dst[12] = f2bf(f3.x); dst[13] = f2bf(f3.y); dst[14] = f2bf(f3.z); dst[15] = f2bf(f3.w);
    }
    wait_async0();
    __syncthreads();
    v16bf af = frag_row(&As[r][0], hs);
    v16bf bw = frag_row(&Bs[wv * 16 + r][0], hs);
    acc = WMMA_BF16(af, bw, acc);
    __syncthreads();
  }
#pragma unroll
  for (int e8 = 0; e8 < 8; e8++) {
    int rl = hs + e8;
    if (mt * 16 + rl < cnt)
      gu[(size_t)(seg + mt * 16 + rl) * (2 * I_) + nt * 64 + wv * 16 + r] = acc[e8];
  }
}

// ---------------------------------------------------------------------------
// MoE down GEMM + top-1 weight scaling + scatter back to token order.
// ---------------------------------------------------------------------------
template <bool WBF16>
__global__ __launch_bounds__(128) void k_moe_down(const bh16* __restrict__ actb,
                                                  const void* __restrict__ ednv,
                                                  const int* __restrict__ sorted,
                                                  const int* __restrict__ offs,
                                                  const float* __restrict__ topw,
                                                  float* __restrict__ moe) {
  int e = blockIdx.z, mt = blockIdx.y, nt = blockIdx.x;
  int seg = offs[e], cnt = offs[e + 1] - seg;
  if (mt * 16 >= cnt) return;
  __shared__ __align__(16) bh16 As[16][40];
  __shared__ __align__(16) bh16 Bs[64][40];
  int tid = threadIdx.x, lane = tid & 31, wv = tid >> 5;
  int r = lane & 15, hs = (lane < 16) ? 0 : 8;

  const bh16*  Wbh = nullptr;
  const float* Wfl = nullptr;
  int lrow = tid >> 1, lhk = (tid & 1) * 16;
  if constexpr (WBF16) Wbh = (const bh16*)ednv + (size_t)e * H_ * I_ + (size_t)(nt * 64 + lrow) * I_ + lhk;
  else                 Wfl = (const float*)ednv + (size_t)e * H_ * I_ + (size_t)(nt * 64 + lrow) * I_ + lhk;

  int arr = (tid >> 2) & 15, asg = (tid & 3) * 8;
  int arow = mt * 16 + arr; if (arow >= cnt) arow = cnt - 1;
  const bh16* Agr = actb + (size_t)(seg + arow) * I_ + asg;
  unsigned ldsAg = lds_off(&As[arr][asg]);
  unsigned ldsB = lds_off(&Bs[lrow][lhk]);

  v8f acc = vzero8();
  for (int k0 = 0; k0 < I_; k0 += 32) {
    if (tid < 64) async_b128(ldsAg, Agr + k0);
    if constexpr (WBF16) {
      async_b128(ldsB,      Wbh + k0);
      async_b128(ldsB + 16, Wbh + k0 + 8);
    } else {
      const float4* s4 = (const float4*)(Wfl + k0);
      float4 f0 = s4[0], f1 = s4[1], f2 = s4[2], f3 = s4[3];
      bh16* dst = &Bs[lrow][lhk];
      dst[0]  = f2bf(f0.x); dst[1]  = f2bf(f0.y); dst[2]  = f2bf(f0.z); dst[3]  = f2bf(f0.w);
      dst[4]  = f2bf(f1.x); dst[5]  = f2bf(f1.y); dst[6]  = f2bf(f1.z); dst[7]  = f2bf(f1.w);
      dst[8]  = f2bf(f2.x); dst[9]  = f2bf(f2.y); dst[10] = f2bf(f2.z); dst[11] = f2bf(f2.w);
      dst[12] = f2bf(f3.x); dst[13] = f2bf(f3.y); dst[14] = f2bf(f3.z); dst[15] = f2bf(f3.w);
    }
    wait_async0();
    __syncthreads();
    v16bf af = frag_row(&As[r][0], hs);
    v16bf bw = frag_row(&Bs[wv * 16 + r][0], hs);
    acc = WMMA_BF16(af, bw, acc);
    __syncthreads();
  }
#pragma unroll
  for (int e8 = 0; e8 < 8; e8++) {
    int rl = hs + e8;
    if (mt * 16 + rl < cnt) {
      int token = sorted[seg + mt * 16 + rl];
      moe[(size_t)token * H_ + nt * 64 + wv * 16 + r] = topw[token] * acc[e8];
    }
  }
}

__global__ __launch_bounds__(256) void k_silu_mul(const float* __restrict__ gu,
                                                  bh16* __restrict__ act,
                                                  int rows, int twoI, int Iw) {
  long i = (long)blockIdx.x * 256 + threadIdx.x;
  long total = (long)rows * Iw;
  if (i < total) {
    long row = i / Iw; int col = (int)(i % Iw);
    float g = gu[row * twoI + col], u = gu[row * twoI + Iw + col];
    float s = g / (1.f + __expf(-g));
    act[i] = f2bf(s * u);
  }
}

__global__ __launch_bounds__(256) void k_final(const float* __restrict__ x2,
                                               const float* __restrict__ moe,
                                               const float* __restrict__ sh,
                                               float* __restrict__ out) {
  long i = (long)blockIdx.x * 256 + threadIdx.x;
  if (i < (long)T_ * H_) out[i] = x2[i] + moe[i] + sh[i];
}

// ---------------------------------------------------------------------------
extern "C" void kernel_launch(void* const* d_in, const int* in_sizes, int n_in,
                              void* d_out, int out_size, void* d_ws, size_t ws_size,
                              hipStream_t stream) {
  const float* x     = (const float*)d_in[0];
  const float* ln1w  = (const float*)d_in[1];
  const float* qkvw  = (const float*)d_in[2];
  const float* qlnw  = (const float*)d_in[3];
  const float* klnw  = (const float*)d_in[4];
  const float* ow    = (const float*)d_in[5];
  const float* ln2w  = (const float*)d_in[6];
  const float* gatew = (const float*)d_in[7];
  const float* egu   = (const float*)d_in[8];
  const float* edn   = (const float*)d_in[9];
  const float* sguw  = (const float*)d_in[10];
  const float* sdnw  = (const float*)d_in[11];

  char* ws = (char*)d_ws;
  const size_t SB = (size_t)T_ * H_ * sizeof(bh16);   // 8 MiB unit
  bh16*  h_bf    = (bh16*)(ws + 0 * SB);
  bh16*  attn_bf = (bh16*)(ws + 1 * SB);
  bh16*  q_bf    = (bh16*)(ws + 2 * SB);
  bh16*  k_bf    = (bh16*)(ws + 3 * SB);
  bh16*  v_bf    = (bh16*)(ws + 4 * SB);
  bh16*  h2_bf   = (bh16*)(ws + 5 * SB);
  float* qkv     = (float*)(ws + 6 * SB);    // 6 SB
  float* x2      = (float*)(ws + 12 * SB);   // 2 SB
  float* gu      = (float*)(ws + 14 * SB);   // 4 SB
  bh16*  act_bf  = (bh16*)(ws + 18 * SB);    // 1 SB
  bh16*  sact_bf = (bh16*)(ws + 19 * SB);    // 1 SB
  char*  small   = ws + 20 * SB;
  int*   topi    = (int*)(small);
  float* topw    = (float*)(small + 8192);
  int*   sorted  = (int*)(small + 16384);
  int*   counts  = (int*)(small + 24576);
  int*   offs    = (int*)(small + 24576 + 128);
  int*   cursor  = (int*)(small + 24576 + 256);
  // Region reuse (stream-serialized):
  float* moe_buf    = (float*)(ws + 0 * SB);  // h_bf+attn_bf (dead by MoE time)
  float* sgu        = (float*)(ws + 6 * SB);  // qkv region (dead after qkvpost)
  float* shared_out = (float*)(ws + 2 * SB);  // q_bf+k_bf (dead after attention)
  // bf16 weight mirrors (tiered on ws_size; deterministic per-harness):
  bh16* qkvw_bf = (bh16*)(ws + 21 * SB);      // 3 SB
  bh16* ow_bf   = (bh16*)(ws + 24 * SB);      // 1 SB
  bh16* sguw_bf = (bh16*)(ws + 25 * SB);      // 2 SB
  bh16* sdnw_bf = (bh16*)(ws + 27 * SB);      // 1 SB
  bh16* egu_bf  = (bh16*)(ws + 28 * SB);      // 32 SB
  bh16* edn_bf  = (bh16*)(ws + 60 * SB);      // 16 SB
  const bool cvt_dense  = ws_size >= 28 * SB;
  const bool cvt_expert = ws_size >= 76 * SB;

  if (cvt_dense) {
    k_cvt<<<(3 * H_ * H_) / 1024, 256, 0, stream>>>(qkvw, qkvw_bf, (long)3 * H_ * H_);
    k_cvt<<<(H_ * H_) / 1024, 256, 0, stream>>>(ow, ow_bf, (long)H_ * H_);
    k_cvt<<<(2 * IS_ * H_) / 1024, 256, 0, stream>>>(sguw, sguw_bf, (long)2 * IS_ * H_);
    k_cvt<<<(H_ * IS_) / 1024, 256, 0, stream>>>(sdnw, sdnw_bf, (long)H_ * IS_);
  }
  if (cvt_expert) {
    k_cvt<<<(E_ * 2 * I_ * H_) / 1024, 256, 0, stream>>>(egu, egu_bf, (long)E_ * 2 * I_ * H_);
    k_cvt<<<(E_ * H_ * I_) / 1024, 256, 0, stream>>>(edn, edn_bf, (long)E_ * H_ * I_);
  }

  k_rms<<<T_, 256, 0, stream>>>(x, ln1w, h_bf);
  if (cvt_dense)
    k_gemm<true><<<dim3((3 * H_) / 64, T_ / 64), 128, 0, stream>>>(h_bf, qkvw_bf, qkv, nullptr, T_, 3 * H_, H_);
  else
    k_gemm<false><<<dim3((3 * H_) / 64, T_ / 64), 128, 0, stream>>>(h_bf, qkvw, qkv, nullptr, T_, 3 * H_, H_);
  k_qkvpost<<<dim3(T_, NH_), 128, 0, stream>>>(qkv, qlnw, klnw, q_bf, k_bf, v_bf);
  k_attn<<<dim3(NH_, T_ / 16), 32, 0, stream>>>(q_bf, k_bf, v_bf, attn_bf);
  if (cvt_dense)
    k_gemm<true><<<dim3(H_ / 64, T_ / 64), 128, 0, stream>>>(attn_bf, ow_bf, x2, x, T_, H_, H_);
  else
    k_gemm<false><<<dim3(H_ / 64, T_ / 64), 128, 0, stream>>>(attn_bf, ow, x2, x, T_, H_, H_);
  k_rms<<<T_, 256, 0, stream>>>(x2, ln2w, h2_bf);
  k_router<<<T_, 128, 0, stream>>>(h2_bf, gatew, topi, topw);
  k_zero16<<<1, 32, 0, stream>>>(counts);
  k_count<<<T_ / 256, 256, 0, stream>>>(topi, counts);
  k_scan<<<1, 32, 0, stream>>>(counts, offs, cursor);
  k_scatter<<<T_ / 256, 256, 0, stream>>>(topi, offs, cursor, sorted);
  if (cvt_expert) {
    k_moe_gu<true><<<dim3((2 * I_) / 64, T_ / 16, E_), 128, 0, stream>>>(h2_bf, egu_bf, sorted, offs, gu);
  } else {
    k_moe_gu<false><<<dim3((2 * I_) / 64, T_ / 16, E_), 128, 0, stream>>>(h2_bf, egu, sorted, offs, gu);
  }
  k_silu_mul<<<(T_ * I_) / 256, 256, 0, stream>>>(gu, act_bf, T_, 2 * I_, I_);
  if (cvt_expert) {
    k_moe_down<true><<<dim3(H_ / 64, T_ / 16, E_), 128, 0, stream>>>(act_bf, edn_bf, sorted, offs, topw, moe_buf);
  } else {
    k_moe_down<false><<<dim3(H_ / 64, T_ / 16, E_), 128, 0, stream>>>(act_bf, edn, sorted, offs, topw, moe_buf);
  }
  if (cvt_dense) {
    k_gemm<true><<<dim3((2 * IS_) / 64, T_ / 64), 128, 0, stream>>>(h2_bf, sguw_bf, sgu, nullptr, T_, 2 * IS_, H_);
  } else {
    k_gemm<false><<<dim3((2 * IS_) / 64, T_ / 64), 128, 0, stream>>>(h2_bf, sguw, sgu, nullptr, T_, 2 * IS_, H_);
  }
  k_silu_mul<<<(T_ * IS_) / 256, 256, 0, stream>>>(sgu, sact_bf, T_, 2 * IS_, IS_);
  if (cvt_dense) {
    k_gemm<true><<<dim3(H_ / 64, T_ / 64), 128, 0, stream>>>(sact_bf, sdnw_bf, shared_out, nullptr, T_, H_, IS_);
  } else {
    k_gemm<false><<<dim3(H_ / 64, T_ / 64), 128, 0, stream>>>(sact_bf, sdnw, shared_out, nullptr, T_, H_, IS_);
  }
  k_final<<<(T_ * H_) / 256, 256, 0, stream>>>(x2, moe_buf, shared_out, (float*)d_out);

  (void)in_sizes; (void)n_in; (void)out_size; (void)ws_size;
}
